// FeatureExtractor_23244363006089
// MI455X (gfx1250) — matrix operand order repaired and checked by reading, not compiled, not run on hardware
//
#include <hip/hip_runtime.h>

// ---------------------------------------------------------------------------
// PV-RCNN style keypoint feature extractor for gfx1250 (MI455X, wave32).
//
// d_in layout (setup_inputs dict order, params flattened in insertion order):
//  [0] keypoints   (4,4096,4)  f32
//  [1] points_xyz  (4,16384,3) f32
//  [2] points_feat (4,16384,1) f32
//  [3] bev_features(4,256,200,176) f32
//  [4..9]   sa[0]: w(4,16) g(16) b(16) w(16,16) g(16) b(16)
//  [10..15] sa[1]: w(4,16) g(16) b(16) w(16,32) g(32) b(32)
//  [16..24] coord: w(3,8) g b  w(8,16) g b  w(16,32) g b
//  [25..27] fusion: w(336,128) g(128) b(128)
// d_out: (16384,128) f32
// ---------------------------------------------------------------------------

typedef __attribute__((ext_vector_type(16))) _Float16 v16h;
typedef __attribute__((ext_vector_type(8)))  _Float16 v8h;
typedef __attribute__((ext_vector_type(8)))  float    v8f;

#define BB       4
#define KK       4096
#define NN       16384
#define MROWS    (BB*KK)        // 16384
#define CB       256
#define HH       200
#define WW       176
#define KC       336            // real fusion K
#define KP       352            // padded fusion K (11 * 32)
#define NKT      11
#define NOUT     128
#define NNT      8              // 128/16
#define TILE     2048           // points staged in LDS per pass
#define SAW_TOT  1056

struct SAPtrs { const float* p[12]; };

// ---------------------------------------------------------------------------
// Kernel 1: BEV bilinear sampling -> A[:, 0:256] (f16)
// ---------------------------------------------------------------------------
__global__ __launch_bounds__(256) void bev_sample_kernel(
    const float* __restrict__ kp, const float* __restrict__ bev,
    _Float16* __restrict__ A)
{
  const int row = blockIdx.x;          // 0..16383
  const int b   = row >> 12;           // /4096
  const int c   = threadIdx.x;         // channel

  const float kx = kp[(size_t)row*4 + 0];
  const float ky = kp[(size_t)row*4 + 1];
  const float x  = kx * 2.5f;                  // (x-0)/0.05/8
  const float y  = (ky + 40.0f) * 2.5f;        // (y+40)/0.05/8

  int x0 = (int)floorf(x), y0 = (int)floorf(y);
  int x1 = x0 + 1, y1 = y0 + 1;
  x0 = x0 < 0 ? 0 : (x0 > WW-1 ? WW-1 : x0);
  x1 = x1 < 0 ? 0 : (x1 > WW-1 ? WW-1 : x1);
  y0 = y0 < 0 ? 0 : (y0 > HH-1 ? HH-1 : y0);
  y1 = y1 < 0 ? 0 : (y1 > HH-1 ? HH-1 : y1);
  const float xf0 = (float)x0, xf1 = (float)x1;
  const float yf0 = (float)y0, yf1 = (float)y1;
  const float wa = (xf1 - x) * (yf1 - y);
  const float wb = (xf1 - x) * (y - yf0);
  const float wc = (x - xf0) * (yf1 - y);
  const float wd = (x - xf0) * (y - yf0);

  const float* base = bev + ((size_t)(b*CB + c)) * (HH*WW);
  const float Ia = base[y0*WW + x0];
  const float Ib = base[y1*WW + x0];
  const float Ic = base[y0*WW + x1];
  const float Id = base[y1*WW + x1];
  const float v  = Ia*wa + Ib*wb + Ic*wc + Id*wd;
  A[(size_t)row*KP + c] = (_Float16)v;
}

// ---------------------------------------------------------------------------
// Kernel 2: pack SA MLP weights contiguously into workspace
// layout (floats): w01@0(64) g01@64 b01@80 w02@96(256) g02@352 b02@368
//                  w11@384(64) g11@448 b11@464 w12@480(512) g12@992 b12@1024
// ---------------------------------------------------------------------------
__global__ void pack_sa_kernel(SAPtrs sp, float* __restrict__ saw)
{
  const int lens[12] = {64,16,16,256,16,16,64,16,16,512,32,32};
  const int offs[12] = {0,64,80,96,352,368,384,448,464,480,992,1024};
  int idx = blockIdx.x * 256 + threadIdx.x;
  if (idx >= SAW_TOT) return;
  for (int s = 0; s < 12; ++s) {
    if (idx >= offs[s] && idx < offs[s] + lens[s]) {
      saw[idx] = sp.p[s][idx - offs[s]];
      return;
    }
  }
}

// ---------------------------------------------------------------------------
// Kernel 3: ball-query + MLP + maxpool, both branches.
// 8 waves/block, one keypoint per wave. Points tiled through LDS.
// Writes A[:, 256:272] (branch 0) and A[:, 272:304] (branch 1).
// ---------------------------------------------------------------------------
__global__ __launch_bounds__(256) void group_pool_kernel(
    const float* __restrict__ kp, const float* __restrict__ pxyz,
    const float* __restrict__ pfeat, const float* __restrict__ saw,
    _Float16* __restrict__ A)
{
  __shared__ float s_x[TILE], s_y[TILE], s_z[TILE], s_f[TILE];
  __shared__ float s_sel0[8][16][4];
  __shared__ float s_sel1[8][32][4];
  __shared__ float s_w[SAW_TOT];

  const int tid  = threadIdx.x;
  const int lane = tid & 31;
  const int w    = tid >> 5;
  const int bpB  = KK / 8;                 // 512 blocks per batch
  const int b    = blockIdx.x / bpB;
  const int k    = (blockIdx.x % bpB) * 8 + w;
  const int row  = b*KK + k;

  for (int i = tid; i < SAW_TOT; i += 256) s_w[i] = saw[i];

  const float kx = kp[(size_t)row*4 + 0];
  const float ky = kp[(size_t)row*4 + 1];
  const float kz = kp[(size_t)row*4 + 2];
  const float* px = pxyz  + (size_t)b*NN*3;
  const float* pf = pfeat + (size_t)b*NN;

  int c1 = 0, c2 = 0;                       // accepted counts (wave-uniform)

  for (int t0 = 0; t0 < NN; t0 += TILE) {
    __syncthreads();
    for (int i = tid; i < TILE; i += 256) {
      const int p = t0 + i;
      s_x[i] = px[(size_t)p*3 + 0];
      s_y[i] = px[(size_t)p*3 + 1];
      s_z[i] = px[(size_t)p*3 + 2];
      s_f[i] = pf[p];
    }
    __syncthreads();

    if (c1 < 16 || c2 < 32) {
      for (int i = lane; i < TILE; i += 32) {
        const float dx = s_x[i] - kx;
        const float dy = s_y[i] - ky;
        const float dz = s_z[i] - kz;
        const float d2 = dx*dx + dy*dy + dz*dz;
        const bool m2 = d2 < 0.64f;               // r=0.8
        const bool m1 = d2 < 0.16f;               // r=0.4
        const unsigned bal1 = (unsigned)__ballot(m1);
        const unsigned bal2 = (unsigned)__ballot(m2);
        const unsigned lt = (1u << lane) - 1u;
        if (m1) {
          const int s = c1 + __popc(bal1 & lt);
          if (s < 16) {
            s_sel0[w][s][0] = dx; s_sel0[w][s][1] = dy;
            s_sel0[w][s][2] = dz; s_sel0[w][s][3] = s_f[i];
          }
        }
        if (m2) {
          const int s = c2 + __popc(bal2 & lt);
          if (s < 32) {
            s_sel1[w][s][0] = dx; s_sel1[w][s][1] = dy;
            s_sel1[w][s][2] = dz; s_sel1[w][s][3] = s_f[i];
          }
        }
        c1 += __popc(bal1);
        c2 += __popc(bal2);
        if (c1 >= 16 && c2 >= 32) break;          // wave-uniform
      }
    }
  }
  __syncthreads();

  // ---- branch 0: 4 -> 16 -> 16, nsample = 16, lane = sample slot ----
  {
    const int cnt = c1 < 16 ? c1 : 16;
    const int s   = (lane < cnt) ? lane : 0;      // duplicates don't change max
    const float g0 = s_sel0[w][s][0], g1v = s_sel0[w][s][1];
    const float g2v = s_sel0[w][s][2], g3v = s_sel0[w][s][3];
    float h1[16];
    #pragma unroll
    for (int h = 0; h < 16; ++h) {
      float sum = g0*s_w[h] + g1v*s_w[16+h] + g2v*s_w[32+h] + g3v*s_w[48+h];
      h1[h] = fmaxf(sum * s_w[64+h] + s_w[80+h], 0.0f);
    }
    float h2[16];
    #pragma unroll
    for (int o = 0; o < 16; ++o) {
      float sum = 0.0f;
      #pragma unroll
      for (int h = 0; h < 16; ++h) sum += h1[h] * s_w[96 + h*16 + o];
      h2[o] = fmaxf(sum * s_w[352+o] + s_w[368+o], 0.0f);
    }
    #pragma unroll
    for (int o = 0; o < 16; ++o) {
      float v = h2[o];
      for (int off = 16; off > 0; off >>= 1) v = fmaxf(v, __shfl_xor(v, off, 32));
      h2[o] = v;
    }
    if (lane == 0) {
      #pragma unroll
      for (int o = 0; o < 16; ++o)
        A[(size_t)row*KP + 256 + o] = (_Float16)((c1 == 0) ? 0.0f : h2[o]);
    }
  }

  // ---- branch 1: 4 -> 16 -> 32, nsample = 32 ----
  {
    const int cnt = c2 < 32 ? c2 : 32;
    const int s   = (lane < cnt) ? lane : 0;
    const float g0 = s_sel1[w][s][0], g1v = s_sel1[w][s][1];
    const float g2v = s_sel1[w][s][2], g3v = s_sel1[w][s][3];
    float h1[16];
    #pragma unroll
    for (int h = 0; h < 16; ++h) {
      float sum = g0*s_w[384+h] + g1v*s_w[384+16+h] + g2v*s_w[384+32+h] + g3v*s_w[384+48+h];
      h1[h] = fmaxf(sum * s_w[448+h] + s_w[464+h], 0.0f);
    }
    float h2[32];
    #pragma unroll
    for (int o = 0; o < 32; ++o) {
      float sum = 0.0f;
      #pragma unroll
      for (int h = 0; h < 16; ++h) sum += h1[h] * s_w[480 + h*32 + o];
      h2[o] = fmaxf(sum * s_w[992+o] + s_w[1024+o], 0.0f);
    }
    #pragma unroll
    for (int o = 0; o < 32; ++o) {
      float v = h2[o];
      for (int off = 16; off > 0; off >>= 1) v = fmaxf(v, __shfl_xor(v, off, 32));
      h2[o] = v;
    }
    if (lane == 0) {
      #pragma unroll
      for (int o = 0; o < 32; ++o)
        A[(size_t)row*KP + 272 + o] = (_Float16)((c2 == 0) ? 0.0f : h2[o]);
    }
  }
}

// ---------------------------------------------------------------------------
// Kernel 4: coord MLP 3->8->16->32 -> A[:, 304:336], zero-pad A[:, 336:352]
// ---------------------------------------------------------------------------
__global__ __launch_bounds__(256) void coord_kernel(
    const float* __restrict__ kp,
    const float* __restrict__ w1, const float* __restrict__ g1, const float* __restrict__ b1,
    const float* __restrict__ w2, const float* __restrict__ g2, const float* __restrict__ b2,
    const float* __restrict__ w3, const float* __restrict__ g3, const float* __restrict__ b3,
    _Float16* __restrict__ A)
{
  const int row = blockIdx.x * 256 + threadIdx.x;
  if (row >= MROWS) return;
  const float x0 = kp[(size_t)row*4+0];
  const float x1 = kp[(size_t)row*4+1];
  const float x2 = kp[(size_t)row*4+2];

  float h1[8];
  #pragma unroll
  for (int j = 0; j < 8; ++j) {
    float s = x0*w1[j] + x1*w1[8+j] + x2*w1[16+j];
    h1[j] = fmaxf(s * g1[j] + b1[j], 0.0f);
  }
  float h2[16];
  #pragma unroll
  for (int j = 0; j < 16; ++j) {
    float s = 0.0f;
    #pragma unroll
    for (int c = 0; c < 8; ++c) s += h1[c] * w2[c*16 + j];
    h2[j] = fmaxf(s * g2[j] + b2[j], 0.0f);
  }
  float h3[32];
  #pragma unroll
  for (int j = 0; j < 32; ++j) {
    float s = 0.0f;
    #pragma unroll
    for (int c = 0; c < 16; ++c) s += h2[c] * w3[c*32 + j];
    h3[j] = fmaxf(s * g3[j] + b3[j], 0.0f);
  }
  _Float16* arow = A + (size_t)row*KP;
  #pragma unroll
  for (int j = 0; j < 32; ++j) arow[304 + j] = (_Float16)h3[j];
  #pragma unroll
  for (int j = 0; j < 16; ++j) arow[336 + j] = (_Float16)0.0f;  // K pad
}

// ---------------------------------------------------------------------------
// Kernel 5: pack fusion weight (336x128 f32) into WMMA B-operand layout, f16.
// Bp[((nt*NKT+kt)*32 + lane)*16 + h] = W[k][n]
//   n = nt*16 + (lane&15);  k = kt*32 + (h<8 ? 8*(lane>>4) + h : 16 + 8*(lane>>4) + (h-8));  zero if k>=336
// ---------------------------------------------------------------------------
__global__ void pack_fw_kernel(const float* __restrict__ fw, _Float16* __restrict__ Bp)
{
  const int idx = blockIdx.x * 256 + threadIdx.x;
  if (idx >= NNT*NKT*32*16) return;
  const int h    = idx & 15;
  const int lane = (idx >> 4) & 31;
  const int kt   = (idx >> 9) % NKT;
  const int nt   = (idx >> 9) / NKT;
  const int n = nt*16 + (lane & 15);
  const int k = kt*32 + ((h < 8) ? 8*(lane >> 4) + h : 16 + 8*(lane >> 4) + (h - 8));
  const float v = (k < KC) ? fw[(size_t)k*NOUT + n] : 0.0f;
  Bp[idx] = (_Float16)v;
}

// ---------------------------------------------------------------------------
// Kernel 6: fusion GEMM via v_wmma_f32_16x16x32_f16.
// One wave per 16x16 output tile; 11 k-tiles of 32. Epilogue: *gamma+beta, relu.
// ---------------------------------------------------------------------------
__global__ __launch_bounds__(256) void fusion_gemm_kernel(
    const _Float16* __restrict__ A, const _Float16* __restrict__ Bp,
    const float* __restrict__ fg, const float* __restrict__ fb,
    float* __restrict__ out)
{
  const int wid  = (blockIdx.x * 256 + threadIdx.x) >> 5;   // 0..8191
  const int lane = threadIdx.x & 31;
  const int mt = wid >> 3;          // 0..1023
  const int nt = wid & 7;           // 0..7
  const int ll = lane & 15;
  const int hi = lane >> 4;         // 0 or 1

  // A operand per ISA 16-bit 16x32 layout: lanes<16 K{0..7,16..23}, lanes>=16 K{8..15,24..31}
  const _Float16* arow = A + (size_t)(mt*16 + ll)*KP + hi*8;
  const _Float16* bcol = Bp + ((size_t)(nt*NKT)*32 + lane)*16;

  v8f acc = {};
  #pragma unroll
  for (int kt = 0; kt < NKT; ++kt) {
    v8h a0 = *(const v8h*)(arow + kt*32);
    v8h a1 = *(const v8h*)(arow + kt*32 + 16);
    v16h av = __builtin_shufflevector(a0, a1, 0,1,2,3,4,5,6,7,8,9,10,11,12,13,14,15);
    v16h bv = *(const v16h*)(bcol + (size_t)kt*32*16);
    acc = __builtin_amdgcn_wmma_f32_16x16x32_f16(
        /*neg_a=*/false, av, /*neg_b=*/false, bv,
        /*c_mod=*/(short)0, acc, /*reuse_a=*/false, /*reuse_b=*/false);
  }

  const int col = nt*16 + ll;
  const float gm = fg[col];
  const float bt = fb[col];
  const int r0 = mt*16 + hi*8;      // C/D layout: VGPR v -> row r0+v, col = lane%16
  #pragma unroll
  for (int v = 0; v < 8; ++v)
    out[(size_t)(r0 + v)*NOUT + col] = fmaxf(acc[v]*gm + bt, 0.0f);
}

// ---------------------------------------------------------------------------
extern "C" void kernel_launch(void* const* d_in, const int* in_sizes, int n_in,
                              void* d_out, int out_size, void* d_ws, size_t ws_size,
                              hipStream_t stream)
{
  (void)in_sizes; (void)n_in; (void)out_size; (void)ws_size;

  const float* kp    = (const float*)d_in[0];
  const float* pxyz  = (const float*)d_in[1];
  const float* pfeat = (const float*)d_in[2];
  const float* bev   = (const float*)d_in[3];

  char* ws = (char*)d_ws;
  _Float16* Abuf = (_Float16*)ws;                               // 16384*352*2 = 11,534,336 B
  _Float16* Bp   = (_Float16*)(ws + 11534336);                  // 45056*2     =     90,112 B (112,640 reserved)
  float*    saw  = (float*)(ws + 11534336 + 112640);            // 1056*4      =      4,224 B

  // (a) BEV bilinear -> A[:,0:256]
  bev_sample_kernel<<<MROWS, 256, 0, stream>>>(kp, bev, Abuf);

  // (b) SA weights pack, then ball-query branches -> A[:,256:304]
  SAPtrs sp;
  for (int i = 0; i < 12; ++i) sp.p[i] = (const float*)d_in[4 + i];
  pack_sa_kernel<<<(SAW_TOT + 255)/256, 256, 0, stream>>>(sp, saw);
  group_pool_kernel<<<BB*(KK/8), 256, 0, stream>>>(kp, pxyz, pfeat, saw, Abuf);

  // (c) coord MLP -> A[:,304:336] + pad -> A[:,336:352]
  coord_kernel<<<MROWS/256, 256, 0, stream>>>(
      kp,
      (const float*)d_in[16], (const float*)d_in[17], (const float*)d_in[18],
      (const float*)d_in[19], (const float*)d_in[20], (const float*)d_in[21],
      (const float*)d_in[22], (const float*)d_in[23], (const float*)d_in[24],
      Abuf);

  // (d) fusion GEMM (WMMA f16 -> f32), scale+bias+relu -> d_out
  pack_fw_kernel<<<(NNT*NKT*32*16 + 255)/256, 256, 0, stream>>>(
      (const float*)d_in[25], Bp);
  fusion_gemm_kernel<<<(MROWS/16)*NNT/8, 256, 0, stream>>>(
      Abuf, Bp, (const float*)d_in[26], (const float*)d_in[27], (float*)d_out);
}
